// SelfAttention_257698038431
// MI455X (gfx1250) — matrix operand order repaired and checked by reading, not compiled, not run on hardware
//
#include <hip/hip_runtime.h>

// ---------------------------------------------------------------------------
// Problem constants (match reference)
// ---------------------------------------------------------------------------
#define B_  2
#define T_  2048
#define C_  2048
#define H_  16
#define KH_ 4
#define D_  128
static constexpr int REP = H_ / KH_;       // 4
static constexpr float EPS_ = 1e-6f;

// ---------------------------------------------------------------------------
// Types for CDNA5 WMMA (wave32). bf16 path: v_wmma_f32_16x16x32_bf16.
// ---------------------------------------------------------------------------
typedef __bf16 bf16;
typedef __attribute__((ext_vector_type(16))) __bf16 v16bf;
typedef __attribute__((ext_vector_type(8)))  __bf16 v8bf;
typedef __attribute__((ext_vector_type(4)))  __bf16 v4bf;
typedef __attribute__((ext_vector_type(8)))  float  v8f;

__device__ __forceinline__ v8f wmma_bf16(v16bf a, v16bf b, v8f c) {
  // (neg_a, A, neg_b, B, c_mod, C, reuse_a, reuse_b)
  return __builtin_amdgcn_wmma_f32_16x16x32_bf16(false, a, false, b,
                                                 (short)0, c, false, false);
}

// --- CDNA5 async global->LDS (ASYNCcnt-tracked, no VGPR staging) -----------
__device__ __forceinline__ void async_load_b128(unsigned lds_off, const void* g) {
  asm volatile("global_load_async_to_lds_b128 %0, %1, off"
               :: "v"(lds_off), "v"((unsigned long long)(uintptr_t)g)
               : "memory");
}
__device__ __forceinline__ void async_load_b128x2(unsigned lds_off, const void* g) {
  // INST_OFFSET is added to BOTH the LDS and the memory address (ISA 08 §4.4)
  asm volatile("global_load_async_to_lds_b128 %0, %1, off\n\t"
               "global_load_async_to_lds_b128 %0, %1, off offset:16"
               :: "v"(lds_off), "v"((unsigned long long)(uintptr_t)g)
               : "memory");
}
__device__ __forceinline__ void wait_async0() {
  asm volatile("s_wait_asynccnt 0x0" ::: "memory");
}

__device__ __forceinline__ float halfmax(float v) {
  v = fmaxf(v, __shfl_xor(v, 1));
  v = fmaxf(v, __shfl_xor(v, 2));
  v = fmaxf(v, __shfl_xor(v, 4));
  v = fmaxf(v, __shfl_xor(v, 8));
  return v;
}
__device__ __forceinline__ float halfsum(float v) {
  v += __shfl_xor(v, 1);
  v += __shfl_xor(v, 2);
  v += __shfl_xor(v, 4);
  v += __shfl_xor(v, 8);
  return v;
}

// ---------------------------------------------------------------------------
// Kernel 1: f32 -> bf16 conversion (vectorized x4)
// ---------------------------------------------------------------------------
__global__ void cvt_f32_bf16(const float* __restrict__ in,
                             bf16* __restrict__ out, int n4) {
  int i = blockIdx.x * blockDim.x + threadIdx.x;
  int stride = gridDim.x * blockDim.x;
  for (; i < n4; i += stride) {
    float4 v = ((const float4*)in)[i];
    v4bf o;
    o[0] = (bf16)v.x; o[1] = (bf16)v.y; o[2] = (bf16)v.z; o[3] = (bf16)v.w;
    ((v4bf*)out)[i] = o;
  }
}

// ---------------------------------------------------------------------------
// Kernel 2: GEMM  out[M,N] = A[M,K](bf16) * W[N,K](bf16)^T, f32 accumulate.
// Workgroup tile 128x128, 8 waves in a 4(m) x 2(n) grid, each wave 32x64.
// K staged in (dynamic) LDS in 32-wide slabs via async global->LDS DMA.
// Dynamic LDS: As @ 0 (8KB), Ws @ 8192 (8KB).
// ---------------------------------------------------------------------------
template <bool STORE_BF16>
__global__ __launch_bounds__(256)
void gemm_nt(const bf16* __restrict__ A, const bf16* __restrict__ W,
             float* __restrict__ outF, bf16* __restrict__ outB,
             int M, int N, int K) {
  extern __shared__ __align__(16) char smem_g[];
  bf16* As = (bf16*)smem_g;                 // [128][32]
  bf16* Ws = (bf16*)(smem_g + 8192);        // [128][32]
  const unsigned lbase = __builtin_amdgcn_groupstaticsize();

  const int tid  = threadIdx.x;
  const int lane = tid & 31;
  const int w    = tid >> 5;
  const int lm   = lane & 15;
  const int hs   = lane >> 4;          // half-wave select (K/M split)
  const int n0   = blockIdx.x * 128;
  const int m0   = blockIdx.y * 128;
  const int wm   = w & 3;              // 0..3  -> rows  [wm*32, +32)
  const int wn   = w >> 2;             // 0..1  -> cols  [wn*64, +64)
  const int arow = tid >> 1;           // 0..127
  const int acol = (tid & 1) * 16;     // 0 or 16 (halves)
  const unsigned aoff = lbase + (unsigned)(arow * 32 + acol) * 2;

  v8f acc[2][4];
  #pragma unroll
  for (int mt = 0; mt < 2; ++mt)
    #pragma unroll
    for (int nt = 0; nt < 4; ++nt)
      #pragma unroll
      for (int r = 0; r < 8; ++r) acc[mt][nt][r] = 0.0f;

  for (int kt = 0; kt < K; kt += 32) {
    // async DMA: each thread moves a 32B strip of A and of W into LDS
    async_load_b128x2(aoff,        A + (size_t)(m0 + arow) * K + kt + acol);
    async_load_b128x2(aoff + 8192, W + (size_t)(n0 + arow) * K + kt + acol);
    wait_async0();
    __syncthreads();

    // A fragments: lane holds row lm; K halves [hs*16, +16) contiguous
    v16bf af[2];
    #pragma unroll
    for (int mt = 0; mt < 2; ++mt) {
      const int ml = wm * 32 + mt * 16 + lm;
      af[mt] = *(const v16bf*)&As[ml * 32 + hs * 16];
    }
    // B fragments: lane holds col lm; K halves [hs*16, +16) contiguous
    v16bf bfr[4];
    #pragma unroll
    for (int nt = 0; nt < 4; ++nt) {
      const int nl = wn * 64 + nt * 16 + lm;
      bfr[nt] = *(const v16bf*)&Ws[nl * 32 + hs * 16];
    }
    #pragma unroll
    for (int mt = 0; mt < 2; ++mt)
      #pragma unroll
      for (int nt = 0; nt < 4; ++nt)
        acc[mt][nt] = wmma_bf16(af[mt], bfr[nt], acc[mt][nt]);
    __syncthreads();
  }

  // C layout: lane lm = col, VGPR r = row r + 8*hs
  #pragma unroll
  for (int mt = 0; mt < 2; ++mt)
    #pragma unroll
    for (int nt = 0; nt < 4; ++nt)
      #pragma unroll
      for (int r = 0; r < 8; ++r) {
        const int m = m0 + wm * 32 + mt * 16 + r + 8 * hs;
        const int n = n0 + wn * 64 + nt * 16 + lm;
        const size_t idx = (size_t)m * N + n;
        if constexpr (STORE_BF16) outB[idx] = (bf16)acc[mt][nt][r];
        else                      outF[idx] = acc[mt][nt][r];
      }
}

// ---------------------------------------------------------------------------
// Kernel 3: RMSNorm + RoPE. One wave per (b,t,head) row of D=128.
// raw layout: [b*T+t, head, d] f32.  Output bf16, same layout.
// ---------------------------------------------------------------------------
__global__ __launch_bounds__(128)
void norm_rope(const float* __restrict__ raw, const float* __restrict__ cosp,
               const float* __restrict__ sinp, const float* __restrict__ wn,
               bf16* __restrict__ outb, int heads, int totalRows) {
  const int gw   = (blockIdx.x * blockDim.x + threadIdx.x) >> 5;
  const int lane = threadIdx.x & 31;
  if (gw >= totalRows) return;
  const int bt = gw / heads;
  const float* r = raw + (size_t)gw * D_;

  float v[4];
  float ss = 0.0f;
  #pragma unroll
  for (int i = 0; i < 4; ++i) { v[i] = r[lane + 32 * i]; ss += v[i] * v[i]; }
  #pragma unroll
  for (int m = 16; m >= 1; m >>= 1) ss += __shfl_xor(ss, m);
  const float inv = rsqrtf(ss * (1.0f / (float)D_) + EPS_);

  float nv[4], cz[4], sz[4];
  #pragma unroll
  for (int i = 0; i < 4; ++i) {
    const int d = lane + 32 * i;
    nv[i] = wn[d] * (v[i] * inv);
    cz[i] = cosp[(size_t)bt * D_ + d];
    sz[i] = sinp[(size_t)bt * D_ + d];
  }
  // rotate_half: d<64 pairs with d+64 (i -> i+2), d>=64 pairs with d-64.
  float o[4];
  o[0] = nv[0] * cz[0] - nv[2] * sz[0];
  o[1] = nv[1] * cz[1] - nv[3] * sz[1];
  o[2] = nv[2] * cz[2] + nv[0] * sz[2];
  o[3] = nv[3] * cz[3] + nv[1] * sz[3];
  #pragma unroll
  for (int i = 0; i < 4; ++i)
    outb[(size_t)gw * D_ + lane + 32 * i] = (bf16)o[i];
}

// ---------------------------------------------------------------------------
// Kernel 4: causal GQA flash attention, bf16 WMMA, f32 softmax state.
// Workgroup = 8 waves x 16 query rows = 128 queries for one (b,h).
// Key blocks of 64 staged in dynamic LDS:
//   Ksh @ 0      (16KB) [key][d]    <- async global->LDS DMA
//   Vt  @ 16384  (16KB) [d][key]    <- register transpose (scatter)
//   Psh @ 32768  (16KB) per-wave 16x64 P strips
// ---------------------------------------------------------------------------
static constexpr int BM = 128;
static constexpr int BN = 64;

__global__ __launch_bounds__(256)
void flash_attn(const bf16* __restrict__ qb, const bf16* __restrict__ kb,
                const bf16* __restrict__ vb, bf16* __restrict__ yb) {
  extern __shared__ __align__(16) char smem_f[];
  bf16* Ksh = (bf16*)smem_f;                  // BN*D_
  bf16* Vt  = (bf16*)(smem_f + 16384);        // D_*BN
  bf16* Psh = (bf16*)(smem_f + 32768);        // 8*16*BN
  const unsigned lbase = __builtin_amdgcn_groupstaticsize();

  const int tid  = threadIdx.x;
  const int lane = tid & 31;
  const int w    = tid >> 5;
  const int lm   = lane & 15;
  const int hs   = lane >> 4;

  const int bid   = blockIdx.x;
  const int qt    = bid % (T_ / BM);
  const int h     = (bid / (T_ / BM)) % H_;
  const int b     = bid / ((T_ / BM) * H_);
  const int kh    = h / REP;
  const int qbase = qt * BM;
  const int row0  = qbase + w * 16;     // wave's first query row
  const int wqmax = row0 + 15;

  // Q fragments (A layout), 4 chunks of K=32 over D=128, held in registers
  v16bf qf[4];
  {
    const bf16* qrow = qb + (((size_t)(b * T_ + row0 + lm)) * H_ + h) * D_;
    #pragma unroll
    for (int dc = 0; dc < 4; ++dc) {
      qf[dc] = *(const v16bf*)(qrow + dc * 32 + hs * 16);
    }
  }

  v8f o[8];
  #pragma unroll
  for (int dt = 0; dt < 8; ++dt)
    #pragma unroll
    for (int r = 0; r < 8; ++r) o[dt][r] = 0.0f;
  float mrow[8], lrow[8];
  #pragma unroll
  for (int r = 0; r < 8; ++r) { mrow[r] = -1e30f; lrow[r] = 0.0f; }

  const float scale = 0.08838834764831845f;   // 1/sqrt(128)
  const int jmax = (qbase + BM - 1) / BN;

  for (int j = 0; j <= jmax; ++j) {
    const int kbase = j * BN;
    // ---- K tile: async DMA straight into LDS (no VGPR staging) ----
    #pragma unroll
    for (int li = 0; li < 4; ++li) {
      const int hf  = li * 2048 + tid * 8;
      const int key = hf >> 7;
      const int d   = hf & 127;
      const bf16* src = kb + (((size_t)(b * T_ + kbase + key)) * KH_ + kh) * D_ + d;
      async_load_b128(lbase + (unsigned)(key * D_ + d) * 2, src);
    }
    // ---- V tile: register transpose into Vt[d][key] ----
    #pragma unroll
    for (int li = 0; li < 4; ++li) {
      const int hf  = li * 2048 + tid * 8;
      const int key = hf >> 7;
      const int d   = hf & 127;
      const bf16* src = vb + (((size_t)(b * T_ + kbase + key)) * KH_ + kh) * D_ + d;
      if (j < jmax)
        __builtin_prefetch(src + (size_t)BN * KH_ * D_, 0, 1);  // global_prefetch_b8
      v8bf tmp = *(const v8bf*)src;
      #pragma unroll
      for (int jj = 0; jj < 8; ++jj) Vt[(d + jj) * BN + key] = tmp[jj];
    }
    wait_async0();
    __syncthreads();

    if (kbase <= wqmax) {
      // ---- S = scale * Q K^T, masked ----
      float st[4][8];
      #pragma unroll
      for (int nt = 0; nt < 4; ++nt) {
        v8f s;
        #pragma unroll
        for (int r = 0; r < 8; ++r) s[r] = 0.0f;
        #pragma unroll
        for (int dc = 0; dc < 4; ++dc) {
          v16bf bfr = *(const v16bf*)&Ksh[(nt * 16 + lm) * D_ + dc * 32 + hs * 16];
          s = wmma_bf16(qf[dc], bfr, s);
        }
        const int key = kbase + nt * 16 + lm;
        #pragma unroll
        for (int r = 0; r < 8; ++r) {
          float val = s[r] * scale;
          const int qr = row0 + r + 8 * hs;
          st[nt][r] = (key > qr) ? -1e30f : val;
        }
      }
      // ---- online softmax ----
      float mnew[8], alpha[8];
      #pragma unroll
      for (int r = 0; r < 8; ++r) {
        float mx = fmaxf(fmaxf(st[0][r], st[1][r]), fmaxf(st[2][r], st[3][r]));
        mx = halfmax(mx);
        mnew[r]  = fmaxf(mrow[r], mx);
        alpha[r] = __expf(mrow[r] - mnew[r]);
        mrow[r]  = mnew[r];
        lrow[r] *= alpha[r];
      }
      #pragma unroll
      for (int dt = 0; dt < 8; ++dt)
        #pragma unroll
        for (int r = 0; r < 8; ++r) o[dt][r] *= alpha[r];

      float psum[8];
      #pragma unroll
      for (int r = 0; r < 8; ++r) psum[r] = 0.0f;
      #pragma unroll
      for (int nt = 0; nt < 4; ++nt)
        #pragma unroll
        for (int r = 0; r < 8; ++r) {
          const float p = __expf(st[nt][r] - mnew[r]);
          psum[r] += p;
          // stage P in per-wave LDS strip (C layout -> A layout transpose)
          Psh[w * (16 * BN) + (r + 8 * hs) * BN + nt * 16 + lm] = (bf16)p;
        }
      #pragma unroll
      for (int r = 0; r < 8; ++r) lrow[r] += halfsum(psum[r]);

      // ---- O += P V  (read P back in A layout; LDS is in-order per wave) ----
      v16bf pf[2];
      #pragma unroll
      for (int kc = 0; kc < 2; ++kc) {
        pf[kc] = *(const v16bf*)&Psh[w * (16 * BN) + lm * BN + kc * 32 + hs * 16];
      }
      #pragma unroll
      for (int dt = 0; dt < 8; ++dt)
        #pragma unroll
        for (int kc = 0; kc < 2; ++kc) {
          v16bf vfr = *(const v16bf*)&Vt[(dt * 16 + lm) * BN + kc * 32 + hs * 16];
          o[dt] = wmma_bf16(pf[kc], vfr, o[dt]);
        }
    }
    __syncthreads();
  }

  // ---- normalize and store y (bf16, [b*T+t, h, d]) ----
  float rcp[8];
  #pragma unroll
  for (int r = 0; r < 8; ++r) rcp[r] = 1.0f / lrow[r];
  #pragma unroll
  for (int dt = 0; dt < 8; ++dt)
    #pragma unroll
    for (int r = 0; r < 8; ++r) {
      const int m = r + 8 * hs;
      const int t = row0 + m;
      yb[(((size_t)(b * T_ + t)) * H_ + h) * D_ + dt * 16 + lm] =
          (bf16)(o[dt][r] * rcp[r]);
    }
}

// ---------------------------------------------------------------------------
// Host-side launch
// ---------------------------------------------------------------------------
extern "C" void kernel_launch(void* const* d_in, const int* in_sizes, int n_in,
                              void* d_out, int out_size, void* d_ws, size_t ws_size,
                              hipStream_t stream) {
  (void)in_sizes; (void)n_in; (void)out_size; (void)ws_size;

  const float* x    = (const float*)d_in[0];
  const float* cosp = (const float*)d_in[1];
  const float* sinp = (const float*)d_in[2];
  const float* Wq   = (const float*)d_in[3];
  const float* Wk   = (const float*)d_in[4];
  const float* Wv   = (const float*)d_in[5];
  const float* Wo   = (const float*)d_in[6];
  const float* qnw  = (const float*)d_in[7];
  const float* knw  = (const float*)d_in[8];
  float* out = (float*)d_out;

  const size_t XB = (size_t)B_ * T_ * C_;        // 8388608
  const size_t WQ = (size_t)H_ * D_ * C_;        // 4194304
  const size_t WK = (size_t)KH_ * D_ * C_;       // 1048576
  const size_t QE = (size_t)B_ * T_ * H_ * D_;   // 8388608
  const size_t KE = (size_t)B_ * T_ * KH_ * D_;  // 2097152

  char* ws = (char*)d_ws;
  size_t off = 0;
  auto alloc = [&](size_t bytes) {
    void* p = ws + off;
    off += (bytes + 255) & ~(size_t)255;
    return p;
  };
  bf16*  xb    = (bf16*)alloc(XB * 2);
  bf16*  wqb   = (bf16*)alloc(WQ * 2);
  bf16*  wkb   = (bf16*)alloc(WK * 2);
  bf16*  wvb   = (bf16*)alloc(WK * 2);
  bf16*  wob   = (bf16*)alloc(WQ * 2);
  float* q_raw = (float*)alloc(QE * 4);
  float* k_raw = (float*)alloc(KE * 4);
  bf16*  qbuf  = (bf16*)alloc(QE * 2);
  bf16*  kbuf  = (bf16*)alloc(KE * 2);
  bf16*  vbuf  = (bf16*)alloc(KE * 2);
  bf16*  ybuf  = (bf16*)alloc(QE * 2);

  // 1) f32 -> bf16 conversions
  auto cvt = [&](const float* src, bf16* dst, size_t n) {
    int n4 = (int)(n / 4);
    int blocks = (n4 + 255) / 256;
    if (blocks > 2048) blocks = 2048;
    cvt_f32_bf16<<<blocks, 256, 0, stream>>>(src, dst, n4);
  };
  cvt(x,  xb,  XB);
  cvt(Wq, wqb, WQ);
  cvt(Wk, wkb, WK);
  cvt(Wv, wvb, WK);
  cvt(Wo, wob, WQ);

  const int M = B_ * T_;        // 4096
  const int GEMM_LDS  = 16384;  // As + Ws
  const int FLASH_LDS = 49152;  // Ksh + Vt + Psh

  // 2) QKV projections (WMMA bf16 -> f32/bf16)
  gemm_nt<false><<<dim3(C_ / 128, M / 128), 256, GEMM_LDS, stream>>>(xb, wqb, q_raw, nullptr, M, C_, C_);
  gemm_nt<false><<<dim3((KH_ * D_) / 128, M / 128), 256, GEMM_LDS, stream>>>(xb, wkb, k_raw, nullptr, M, KH_ * D_, C_);
  gemm_nt<true ><<<dim3((KH_ * D_) / 128, M / 128), 256, GEMM_LDS, stream>>>(xb, wvb, nullptr, vbuf, M, KH_ * D_, C_);

  // 3) RMSNorm + RoPE on q and k
  {
    const int qrows = M * H_;               // 65536 rows, 4 waves/block
    norm_rope<<<qrows / 4, 128, 0, stream>>>(q_raw, cosp, sinp, qnw, qbuf, H_, qrows);
    const int krows = M * KH_;              // 16384 rows
    norm_rope<<<krows / 4, 128, 0, stream>>>(k_raw, cosp, sinp, knw, kbuf, KH_, krows);
  }

  // 4) causal GQA flash attention
  flash_attn<<<B_ * H_ * (T_ / BM), 256, FLASH_LDS, stream>>>(qbuf, kbuf, vbuf, ybuf);

  // 5) output projection -> f32 d_out
  gemm_nt<false><<<dim3(C_ / 128, M / 128), 256, GEMM_LDS, stream>>>(ybuf, wob, out, nullptr, M, C_, C_);
}
